// HyKT_37391985279186
// MI455X (gfx1250) — matrix operand order repaired and checked by reading, not compiled, not run on hardware
//
#include <hip/hip_runtime.h>
#include <cstdint>

// ---------------- model constants ----------------
#define DD    128
#define BSZ   256
#define SSZ   400
#define NV    11965
#define NHE   189
#define MEMB  3
#define NPAIR (NV*MEMB)
#define BS    (BSZ*SSZ)
#define BN_EPS 1e-5f

typedef __attribute__((ext_vector_type(16))) __bf16 v16bf;
typedef __attribute__((ext_vector_type(8)))  float  v8f;

static __device__ __forceinline__ float sigm(float v) { return 1.0f / (1.0f + expf(-v)); }

union AFrag { v16bf v; __bf16 h[16]; };
union BFrag { v16bf v; uint4 q[2]; };
union BF1   { __bf16 h; unsigned short u; };

static __device__ __forceinline__ unsigned short f2bf_bits(float f) {
  BF1 b; b.h = (__bf16)f;      // native RNE convert on gfx1250
  return b.u;
}

// A fragment for v_wmma_*_16x16x32_bf16: lane l holds row (l&15);
// lanes<16: K = {0..7, 16..23}+kt*32 ; lanes>=16: K = {8..15, 24..31}+kt*32.
// ar must already point at row + kt*32 + ((lane&16)?8:0).
static __device__ __forceinline__ v16bf load_afrag(const float* __restrict__ ar) {
  float4 r0 = ((const float4*)ar)[0];
  float4 r1 = ((const float4*)ar)[1];
  float4 r2 = ((const float4*)(ar + 16))[0];
  float4 r3 = ((const float4*)(ar + 16))[1];
  AFrag a;
  a.h[0]=(__bf16)r0.x;  a.h[1]=(__bf16)r0.y;  a.h[2]=(__bf16)r0.z;  a.h[3]=(__bf16)r0.w;
  a.h[4]=(__bf16)r1.x;  a.h[5]=(__bf16)r1.y;  a.h[6]=(__bf16)r1.z;  a.h[7]=(__bf16)r1.w;
  a.h[8]=(__bf16)r2.x;  a.h[9]=(__bf16)r2.y;  a.h[10]=(__bf16)r2.z; a.h[11]=(__bf16)r2.w;
  a.h[12]=(__bf16)r3.x; a.h[13]=(__bf16)r3.y; a.h[14]=(__bf16)r3.z; a.h[15]=(__bf16)r3.w;
  return a.v;
}
// B fragments pre-packed: 32x16 block -> 512 bf16; lane reads 32 contiguous bytes.
static __device__ __forceinline__ v16bf load_bfrag(const unsigned short* __restrict__ Wp,
                                                   int blk, int lane) {
  BFrag b;
  const uint4* p = (const uint4*)Wp + (size_t)blk * 64 + lane * 2;
  b.q[0] = p[0]; b.q[1] = p[1];
  return b.v;
}

// ---- pack W[K,N] f32 row-major into B-fragment-major bf16 (ISA 7.12.2 layout) ----
__global__ void hykt_pack(const float* __restrict__ W, unsigned short* __restrict__ Wp,
                          int K, int N) {
  int t = blockIdx.x * blockDim.x + threadIdx.x;
  if (t >= K * N) return;
  int blk = t >> 9, r = t & 511;
  int lane = r >> 4, idx = r & 15;
  int v = idx >> 1, half = idx & 1;
  int ntiles = N >> 4;
  int kt = blk / ntiles, nt = blk % ntiles;
  int k = kt * 32 + ((lane & 16) ? 8 : 0) + ((idx & 8) ? 16 : 0) + (idx & 7);
  int n = nt * 16 + (lane & 15);
  Wp[t] = f2bf_bits(W[(size_t)k * N + n]);
}

// ---- generic WMMA GEMM: C = act( (A@W) * alpha + beta + rowbias[rowsel[row]] ) ----
// block = 128 threads = 4 waves; each wave: 16 rows x 128 cols; grid.y = N/128.
__global__ void hykt_gemm(const float* __restrict__ A, int lda,
                          const unsigned short* __restrict__ Wp,
                          int M, int K, int N,
                          const float* __restrict__ alpha, const float* __restrict__ beta,
                          const float* __restrict__ rowbias, const int* __restrict__ rowsel,
                          int act, float* __restrict__ C, int ldc) {
  int lane = threadIdx.x & 31;
  int wv = threadIdx.x >> 5;
  int row0 = (blockIdx.x * 4 + wv) * 16;
  if (row0 >= M) return;
  int colbase = blockIdx.y * 128;
  int arow = min(row0 + (lane & 15), M - 1);
  const float* abase = A + (size_t)arow * lda + ((lane & 16) ? 8 : 0);
  v8f acc[8] = {};
  int ntiles = N >> 4;
  int kts = K >> 5;
  for (int kt = 0; kt < kts; ++kt) {
    v16bf af = load_afrag(abase + kt * 32);
#pragma unroll
    for (int nt = 0; nt < 8; ++nt) {
      int blk = kt * ntiles + (colbase >> 4) + nt;
      v16bf bf = load_bfrag(Wp, blk, lane);
      acc[nt] = __builtin_amdgcn_wmma_f32_16x16x32_bf16(false, af, false, bf,
                                                        (short)0, acc[nt], false, false);
    }
  }
  int rbase = row0 + ((lane & 16) ? 8 : 0);
#pragma unroll
  for (int nt = 0; nt < 8; ++nt) {
    int col = colbase + nt * 16 + (lane & 15);
#pragma unroll
    for (int j = 0; j < 8; ++j) {
      int row = rbase + j;
      if (row < M) {
        float v = acc[nt][j];
        if (alpha)   v = v * alpha[col] + beta[col];
        if (rowbias) v += rowbias[(size_t)rowsel[row] * N + col];
        if (act == 1) v = tanhf(v);
        C[(size_t)row * ldc + col] = v;
      }
    }
  }
}

// ---- gathered-A GEMM: x = tanh(concat(e,ed,ep,c,cd)[row] @ Wi + bi), K=640, N=128 ----
__global__ void hykt_xgemm(const float* __restrict__ X2,
                           const float* __restrict__ ed_t, const float* __restrict__ ep_t,
                           const float* __restrict__ c_t,  const float* __restrict__ cd_t,
                           const int* __restrict__ in_e,  const int* __restrict__ in_ed,
                           const int* __restrict__ in_ep, const int* __restrict__ ec_map,
                           const int* __restrict__ ecd_map,
                           const unsigned short* __restrict__ Wpi,
                           const float* __restrict__ bi, float* __restrict__ Xout) {
  int lane = threadIdx.x & 31;
  int wv = threadIdx.x >> 5;
  int row0 = (blockIdx.x * 4 + wv) * 16;          // BS % 64 == 0 -> always in range
  int r = row0 + (lane & 15);
  int e = in_e[r];
  const float* segp[5];
  segp[0] = X2   + (size_t)e * DD;
  segp[1] = ed_t + (size_t)in_ed[r] * DD;
  segp[2] = ep_t + (size_t)in_ep[r] * DD;
  segp[3] = c_t  + (size_t)ec_map[e] * DD;
  segp[4] = cd_t + (size_t)ecd_map[e] * DD;
  int laneoff = (lane & 16) ? 8 : 0;
  v8f acc[8] = {};
  for (int kt = 0; kt < 20; ++kt) {              // K = 640
    const float* ar = segp[kt >> 2] + (kt & 3) * 32 + laneoff;
    v16bf af = load_afrag(ar);
#pragma unroll
    for (int nt = 0; nt < 8; ++nt) {
      v16bf bf = load_bfrag(Wpi, kt * 8 + nt, lane);
      acc[nt] = __builtin_amdgcn_wmma_f32_16x16x32_bf16(false, af, false, bf,
                                                        (short)0, acc[nt], false, false);
    }
  }
  int rbase = row0 + laneoff;
#pragma unroll
  for (int nt = 0; nt < 8; ++nt) {
    int col = nt * 16 + (lane & 15);
#pragma unroll
    for (int j = 0; j < 8; ++j)
      Xout[(size_t)(rbase + j) * DD + col] = tanhf(acc[nt][j] + bi[col]);
  }
}

// ---- small helper kernels ----
__global__ void hykt_zero(float* __restrict__ p, int n) {
  int t = blockIdx.x * blockDim.x + threadIdx.x;
  if (t < n) p[t] = 0.0f;
}
__global__ void hykt_bnprep(const float* b, const float* g, const float* be,
                            const float* m, const float* v,
                            float* alpha, float* beta) {
  int i = threadIdx.x;
  float a = g[i] * rsqrtf(v[i] + BN_EPS);
  alpha[i] = a;
  beta[i]  = (b[i] - m[i]) * a + be[i];
}
__global__ void hykt_edeg(const int* __restrict__ pair_e, float* __restrict__ edeg) {
  int t = blockIdx.x * blockDim.x + threadIdx.x;
  if (t < NPAIR) atomicAdd(&edeg[pair_e[t]], 1.0f);
}
__global__ void hykt_edgeacc(const float* __restrict__ Y, const int* __restrict__ pair_v,
                             const int* __restrict__ pair_e, float* __restrict__ E) {
  int t = blockIdx.x * blockDim.x + threadIdx.x;
  if (t >= NPAIR * DD) return;
  int p = t >> 7, d = t & 127;
  atomicAdd(&E[(size_t)pair_e[p] * DD + d], Y[(size_t)pair_v[p] * DD + d]);
}
__global__ void hykt_ediv(float* __restrict__ E, const float* __restrict__ edeg) {
  int t = blockIdx.x * blockDim.x + threadIdx.x;
  if (t >= NHE * DD) return;
  E[t] /= fmaxf(edeg[t >> 7], 1.0f);
}
__global__ void hykt_v2v(const float* __restrict__ E, const int* __restrict__ pair_e,
                         float* __restrict__ X, int relu) {
  int t = blockIdx.x * blockDim.x + threadIdx.x;
  if (t >= NV * DD) return;
  int v = t >> 7, d = t & 127;
  int b = v * MEMB;
  float s = E[(size_t)pair_e[b] * DD + d] + E[(size_t)pair_e[b + 1] * DD + d] +
            E[(size_t)pair_e[b + 2] * DD + d];
  s *= (1.0f / (float)MEMB);                 // v_deg == MEMB by construction
  X[t] = relu ? fmaxf(s, 0.0f) : s;
}
// avec[a][c] = a_table[a] @ Wg[128:256, c] + bg[c]   (only 2 possible a values)
__global__ void hykt_avec(const float* __restrict__ a_table, const float* __restrict__ Wg,
                          const float* __restrict__ bg, float* __restrict__ avec) {
  int t = blockIdx.x * blockDim.x + threadIdx.x;
  if (t >= 2 * 384) return;
  int a = t / 384, c = t % 384;
  float s = bg[c];
  for (int d = 0; d < DD; ++d) s += a_table[a * DD + d] * Wg[(size_t)(DD + d) * 384 + c];
  avec[t] = s;
}
// per-step GRU elementwise + prediction head: 1 block per batch row
__global__ void hykt_gru_step(int s, const float* __restrict__ x,
                              const float* __restrict__ gi, const float* __restrict__ gh,
                              float* __restrict__ h, const float* __restrict__ We,
                              const float* __restrict__ beo, float* __restrict__ out_main) {
  int b = blockIdx.x, d = threadIdx.x;
  float h_old = h[(size_t)b * DD + d];
  float xv = x[((size_t)b * SSZ + s) * DD + d];
  __shared__ float red[DD];
  red[d] = h_old * xv * We[d];
  __syncthreads();
  for (int off = 64; off > 0; off >>= 1) {
    if (d < off) red[d] += red[d + off];
    __syncthreads();
  }
  if (d == 0) out_main[(size_t)b * SSZ + s] = sigm(red[0] + beo[0]);
  const float* gir = gi + ((size_t)b * SSZ + s) * 384;
  const float* ghr = gh + (size_t)b * 384;
  float z = sigm(gir[d] + ghr[d]);
  float r = sigm(gir[128 + d] + ghr[128 + d]);
  float n = tanhf(gir[256 + d] + r * ghr[256 + d]);
  h[(size_t)b * DD + d] = (1.0f - z) * n + z * h_old;
}
// pred_s = sigmoid(concat(ha,ca,as,it) @ Ws + bs)
__global__ void hykt_preds(const int* __restrict__ ha_i, const int* __restrict__ ca_i,
                           const int* __restrict__ as_i, const int* __restrict__ it_i,
                           const float* __restrict__ ha_t, const float* __restrict__ ca_t,
                           const float* __restrict__ as_t, const float* __restrict__ it_t,
                           const float* __restrict__ Ws, const float* __restrict__ bs,
                           float* __restrict__ out) {
  int t = blockIdx.x * blockDim.x + threadIdx.x;
  if (t >= BS) return;
  const float* h1 = ha_t + (size_t)ha_i[t] * DD;
  const float* h2 = ca_t + (size_t)ca_i[t] * DD;
  const float* h3 = as_t + (size_t)as_i[t] * DD;
  const float* h4 = it_t + (size_t)it_i[t] * DD;
  float acc = bs[0];
  for (int j = 0; j < DD; ++j)
    acc += h1[j] * Ws[j] + h2[j] * Ws[DD + j] + h3[j] * Ws[2 * DD + j] + h4[j] * Ws[3 * DD + j];
  out[t] = sigm(acc);
}

extern "C" void kernel_launch(void* const* d_in, const int* in_sizes, int n_in,
                              void* d_out, int out_size, void* d_ws, size_t ws_size,
                              hipStream_t stream) {
  (void)in_sizes; (void)n_in; (void)out_size; (void)ws_size;
  // -------- inputs --------
  const int*   in_e   = (const int*)d_in[0];
  const int*   in_ed  = (const int*)d_in[1];
  const int*   in_ep  = (const int*)d_in[2];
  const int*   in_a   = (const int*)d_in[3];
  const int*   in_as  = (const int*)d_in[4];
  const int*   in_ha  = (const int*)d_in[5];
  const int*   in_ca  = (const int*)d_in[6];
  const int*   in_it  = (const int*)d_in[7];
  const float* deliver_h = (const float*)d_in[8];
  const int*   pair_v = (const int*)d_in[9];
  const int*   pair_e = (const int*)d_in[10];
  const int*   ec_map = (const int*)d_in[11];
  const int*   ecd_map= (const int*)d_in[12];
  const float* e_table = (const float*)d_in[13];
  const float* c_table = (const float*)d_in[14];
  const float* ed_table= (const float*)d_in[15];
  const float* cd_table= (const float*)d_in[16];
  const float* a_table = (const float*)d_in[17];
  const float* it_table= (const float*)d_in[18];
  const float* ep_table= (const float*)d_in[19];
  const float* ha_table= (const float*)d_in[20];
  const float* ca_table= (const float*)d_in[21];
  const float* as_table= (const float*)d_in[22];
  const float* W1=(const float*)d_in[23]; const float* b1=(const float*)d_in[24];
  const float* g1=(const float*)d_in[25]; const float* be1=(const float*)d_in[26];
  const float* m1=(const float*)d_in[27]; const float* v1=(const float*)d_in[28];
  const float* W2=(const float*)d_in[29]; const float* b2=(const float*)d_in[30];
  const float* g2=(const float*)d_in[31]; const float* be2=(const float*)d_in[32];
  const float* m2=(const float*)d_in[33]; const float* v2=(const float*)d_in[34];
  const float* Wi=(const float*)d_in[35]; const float* bi=(const float*)d_in[36];
  const float* Wg=(const float*)d_in[37]; const float* Ug=(const float*)d_in[38];
  const float* bg=(const float*)d_in[39];
  const float* We=(const float*)d_in[40]; const float* beo=(const float*)d_in[41];
  const float* Ws=(const float*)d_in[42]; const float* bs=(const float*)d_in[43];

  float* out_s    = (float*)d_out;         // pred_s  [B,S]
  float* out_main = (float*)d_out + BS;    // pred_main [B,S]

  // -------- workspace carve --------
  char* p = (char*)d_ws;
  auto alloc = [&](size_t bytes) -> void* {
    void* r = (void*)p;
    p += (bytes + 255) & ~(size_t)255;
    return r;
  };
  unsigned short* Wp1 = (unsigned short*)alloc(128 * 128 * 2);
  unsigned short* Wp2 = (unsigned short*)alloc(128 * 128 * 2);
  unsigned short* Wpi = (unsigned short*)alloc(640 * 128 * 2);
  unsigned short* Wpg = (unsigned short*)alloc(128 * 384 * 2);
  unsigned short* Wpu = (unsigned short*)alloc(128 * 384 * 2);
  float* alpha1 = (float*)alloc(128 * 4);  float* beta1 = (float*)alloc(128 * 4);
  float* alpha2 = (float*)alloc(128 * 4);  float* beta2 = (float*)alloc(128 * 4);
  float* Y    = (float*)alloc((size_t)NV * DD * 4);
  float* Ebuf = (float*)alloc((size_t)NHE * DD * 4);
  float* edeg = (float*)alloc((size_t)NHE * 4);
  float* X1   = (float*)alloc((size_t)NV * DD * 4);
  float* X2   = (float*)alloc((size_t)NV * DD * 4);
  float* xbuf = (float*)alloc((size_t)BS * DD * 4);
  float* avec = (float*)alloc(2 * 384 * 4);
  float* gi   = (float*)alloc((size_t)BS * 384 * 4);
  float* gh   = (float*)alloc((size_t)BSZ * 384 * 4);
  float* h    = (float*)alloc((size_t)BSZ * DD * 4);

  // -------- pack weights to bf16 B-fragments --------
  hykt_pack<<<(128 * 128 + 255) / 256, 256, 0, stream>>>(W1, Wp1, 128, 128);
  hykt_pack<<<(128 * 128 + 255) / 256, 256, 0, stream>>>(W2, Wp2, 128, 128);
  hykt_pack<<<(640 * 128 + 255) / 256, 256, 0, stream>>>(Wi, Wpi, 640, 128);
  hykt_pack<<<(128 * 384 + 255) / 256, 256, 0, stream>>>(Wg, Wpg, 128, 384); // top 128 rows
  hykt_pack<<<(128 * 384 + 255) / 256, 256, 0, stream>>>(Ug, Wpu, 128, 384);
  hykt_bnprep<<<1, 128, 0, stream>>>(b1, g1, be1, m1, v1, alpha1, beta1);
  hykt_bnprep<<<1, 128, 0, stream>>>(b2, g2, be2, m2, v2, alpha2, beta2);

  // -------- HGNN+ conv 1 --------
  hykt_gemm<<<dim3((NV + 63) / 64, 1), 128, 0, stream>>>(
      e_table, DD, Wp1, NV, DD, DD, alpha1, beta1, nullptr, nullptr, 0, Y, DD);
  hykt_zero<<<(NHE + 255) / 256, 256, 0, stream>>>(edeg, NHE);
  hykt_zero<<<(NHE * DD + 255) / 256, 256, 0, stream>>>(Ebuf, NHE * DD);
  hykt_edeg<<<(NPAIR + 255) / 256, 256, 0, stream>>>(pair_e, edeg);
  hykt_edgeacc<<<(NPAIR * DD + 255) / 256, 256, 0, stream>>>(Y, pair_v, pair_e, Ebuf);
  hykt_ediv<<<(NHE * DD + 255) / 256, 256, 0, stream>>>(Ebuf, edeg);
  hykt_v2v<<<(NV * DD + 255) / 256, 256, 0, stream>>>(Ebuf, pair_e, X1, 1);

  // -------- HGNN+ conv 2 --------
  hykt_gemm<<<dim3((NV + 63) / 64, 1), 128, 0, stream>>>(
      X1, DD, Wp2, NV, DD, DD, alpha2, beta2, nullptr, nullptr, 0, Y, DD);
  hykt_zero<<<(NHE + 255) / 256, 256, 0, stream>>>(edeg, NHE);
  hykt_zero<<<(NHE * DD + 255) / 256, 256, 0, stream>>>(Ebuf, NHE * DD);
  hykt_edeg<<<(NPAIR + 255) / 256, 256, 0, stream>>>(pair_e, edeg);
  hykt_edgeacc<<<(NPAIR * DD + 255) / 256, 256, 0, stream>>>(Y, pair_v, pair_e, Ebuf);
  hykt_ediv<<<(NHE * DD + 255) / 256, 256, 0, stream>>>(Ebuf, edeg);
  hykt_v2v<<<(NV * DD + 255) / 256, 256, 0, stream>>>(Ebuf, pair_e, X2, 0);

  // -------- x = tanh(concat @ Wi + bi), gathered A, K=640 --------
  hykt_xgemm<<<dim3(BS / 64, 1), 128, 0, stream>>>(
      X2, ed_table, ep_table, c_table, cd_table,
      in_e, in_ed, in_ep, ec_map, ecd_map, Wpi, bi, xbuf);

  // -------- hoisted GRU input gates: gi = x @ Wg[:128] + avec[a] --------
  hykt_avec<<<(2 * 384 + 255) / 256, 256, 0, stream>>>(a_table, Wg, bg, avec);
  hykt_gemm<<<dim3(BS / 64, 3), 128, 0, stream>>>(
      xbuf, DD, Wpg, BS, DD, 384, nullptr, nullptr, avec, in_a, 0, gi, 384);

  // -------- sequential GRU scan (400 steps): gh = h @ Ug, then gates --------
  hipMemcpyAsync(h, deliver_h, (size_t)BSZ * DD * 4, hipMemcpyDeviceToDevice, stream);
  for (int s = 0; s < SSZ; ++s) {
    hykt_gemm<<<dim3(4, 3), 128, 0, stream>>>(
        h, DD, Wpu, BSZ, DD, 384, nullptr, nullptr, nullptr, nullptr, 0, gh, 384);
    hykt_gru_step<<<BSZ, DD, 0, stream>>>(s, xbuf, gi, gh, h, We, beo, out_main);
  }

  // -------- pred_s head --------
  hykt_preds<<<(BS + 255) / 256, 256, 0, stream>>>(
      in_ha, in_ca, in_as, in_it, ha_table, ca_table, as_table, it_table, Ws, bs, out_s);
}
